// SoftgroupAttention_11836929868321
// MI455X (gfx1250) — matrix operand order repaired and checked by reading, not compiled, not run on hardware
//
#include <hip/hip_runtime.h>
#include <hip/hip_bf16.h>

// ---------------------------------------------------------------------------
// SoftgroupAttention for MI455X (gfx1250, wave32, WMMA bf16 16x16x32)
//
//   1) qkv_gemm   : x(8192x384) @ W_qkv(384x1152) -> Q,K,V bf16 (b,h,n,32)
//   2) gw_kernel  : gw = softmax(q . gp / softplus(t)), padded 20->32 cols
//   3) colsum_pass: WMMA QK^T + WMMA gw gw^T, flash row stats, column sums
//   4) vscale     : V' = V / (colsum + 1e-8), stored TRANSPOSED (bh, d, m)
//   5) out_pass   : recompute P = softmax(QK^T)*GW, P @ V' via WMMA
//   6) proj_gemm  : attn_out(8192x384) @ W_proj(384x384) -> f32 out
//
// All WMMA operands are 16-byte vector loads (global_load_b128 /
// ds_load_b128); weight tiles live transposed in LDS so the B operand's
// K-axis is contiguous per lane. Tile staging uses float4 global loads and
// packed bf16 LDS stores with compile-time trip counts.
// ---------------------------------------------------------------------------

typedef __attribute__((ext_vector_type(16))) __bf16 v16bf;
typedef __attribute__((ext_vector_type(8)))  __bf16 v8bf;
typedef __attribute__((ext_vector_type(4)))  __bf16 v4bf;
typedef __attribute__((ext_vector_type(4)))  float  v4f;
typedef __attribute__((ext_vector_type(8)))  float  v8f;

#define NB   8
#define NH   12
#define NN   1024
#define HD   32
#define DIMC 384
#define QKVC 1152
#define ROWS (NB * NN)          // 8192
#define BHT  (NB * NH)          // 96
#define LDP  40                 // LDS row pad: 80B rows, 16B-aligned subtiles

__device__ __forceinline__ v8f wmma_bf16(v16bf a, v16bf b, v8f c) {
  return __builtin_amdgcn_wmma_f32_16x16x32_bf16(false, a, false, b,
                                                 (short)0, c, false, false);
}
__device__ __forceinline__ v16bf cat8(v8bf a, v8bf b) {
  return __builtin_shufflevector(a, b, 0, 1, 2, 3, 4, 5, 6, 7,
                                 8, 9, 10, 11, 12, 13, 14, 15);
}
__device__ __forceinline__ v8bf ldg8(const __bf16* p) {
  return *(const v8bf*)p;     // 16B aligned by construction -> b128
}
__device__ __forceinline__ v4bf cvt4(v4f f) {
  v4bf h;
  h[0] = (__bf16)f[0]; h[1] = (__bf16)f[1];
  h[2] = (__bf16)f[2]; h[3] = (__bf16)f[3];
  return h;
}

// ---------------------------------------------------------------------------
// Kernel 1: QKV projection. Block 128(M) x 64(N), 8 waves (4x2).
// ---------------------------------------------------------------------------
__global__ __launch_bounds__(256)
void sga_qkv_gemm(const float* __restrict__ x, const float* __restrict__ Wqkv,
                  __bf16* __restrict__ qo, __bf16* __restrict__ ko,
                  __bf16* __restrict__ vo) {
  __shared__ __bf16 xa[128][LDP];    // A tile: [row][k]
  __shared__ __bf16 wbT[64][LDP];    // B tile TRANSPOSED: [col][k]
  const int tid  = threadIdx.x;
  const int wave = tid >> 5, lane = tid & 31;
  const int waveM = wave >> 1, waveN = wave & 1;
  const int bm = blockIdx.x * 128, bn = blockIdx.y * 64;
  const int lhalf = lane >> 4, lmod = lane & 15;
  v8f acc[2][2] = {};

  for (int k0 = 0; k0 < DIMC; k0 += 32) {
    // A tile: 128x32 f32 -> bf16; float4 loads, b64 LDS stores. 1024 chunks.
    #pragma unroll
    for (int it = 0; it < 4; ++it) {
      const int i = tid + it * 256;
      const int r = i >> 3, c4 = (i & 7) * 4;
      v4f f = *(const v4f*)&x[(size_t)(bm + r) * DIMC + k0 + c4];
      *(v4bf*)&xa[r][c4] = cvt4(f);
    }
    // B tile transposed: 32x64 f32; float4 loads, 4 scalar b16 stores. 512.
    #pragma unroll
    for (int it = 0; it < 2; ++it) {
      const int i = tid + it * 256;
      const int r = i >> 4, c4 = (i & 15) * 4;   // r = k, c4 = col
      v4f f = *(const v4f*)&Wqkv[(size_t)(k0 + r) * QKVC + bn + c4];
      wbT[c4 + 0][r] = (__bf16)f[0];
      wbT[c4 + 1][r] = (__bf16)f[1];
      wbT[c4 + 2][r] = (__bf16)f[2];
      wbT[c4 + 3][r] = (__bf16)f[3];
    }
    if (k0 + 32 < DIMC) {            // prefetch next K tile
      __builtin_prefetch(&x[(size_t)(bm + (tid >> 1)) * DIMC + k0 + 32], 0, 1);
      __builtin_prefetch(&Wqkv[(size_t)(k0 + 32 + (tid >> 3)) * QKVC + bn], 0, 1);
    }
    __syncthreads();
    #pragma unroll
    for (int tm = 0; tm < 2; ++tm) {
      const int row = waveM * 32 + tm * 16 + lmod;
      const int kb = lhalf * 16;
      v16bf a = cat8(*(const v8bf*)&xa[row][kb],
                     *(const v8bf*)&xa[row][kb + 8]);
      #pragma unroll
      for (int tn = 0; tn < 2; ++tn) {
        const int col = waveN * 32 + tn * 16 + lmod;
        v16bf b = cat8(*(const v8bf*)&wbT[col][lhalf * 16],
                       *(const v8bf*)&wbT[col][lhalf * 16 + 8]);
        acc[tm][tn] = wmma_bf16(a, b, acc[tm][tn]);
      }
    }
    __syncthreads();
  }
  // scatter to (b, head, n, d): tensor select is uniform per block
  const int which = blockIdx.y / 6;              // 6 x 64-col tiles per tensor
  const int bnrem = bn - which * DIMC;
  __bf16* const dst = (which == 0) ? qo : (which == 1) ? ko : vo;
  #pragma unroll
  for (int tm = 0; tm < 2; ++tm)
    #pragma unroll
    for (int tn = 0; tn < 2; ++tn) {
      const int c2 = bnrem + waveN * 32 + tn * 16 + lmod;
      const int head = c2 >> 5, d = c2 & 31;
      #pragma unroll
      for (int r = 0; r < 8; ++r) {
        const int row = bm + waveM * 32 + tm * 16 + r + lhalf * 8;
        const int b = row >> 10, n = row & 1023;
        dst[(((size_t)b * NH + head) * NN + n) * HD + d] = (__bf16)acc[tm][tn][r];
      }
    }
}

// ---------------------------------------------------------------------------
// Kernel 2: group weights gw = softmax(q . gp / softplus(t)), pad 20 -> 32
// ---------------------------------------------------------------------------
__global__ __launch_bounds__(256)
void sga_gw(const __bf16* __restrict__ q, const float* __restrict__ gp_w,
            const float* __restrict__ t_param, __bf16* __restrict__ gw_pad) {
  const int idx = blockIdx.x * 256 + threadIdx.x;   // (b*NH+h)*NN + n
  if (idx >= BHT * NN) return;
  const int h = (idx >> 10) % NH;
  const float t = log1pf(__expf(t_param[0]));        // softplus
  const float invt = 1.0f / t;
  float qrow[HD];
  #pragma unroll
  for (int d8 = 0; d8 < HD; d8 += 8) {
    v8bf qv = ldg8(q + (size_t)idx * HD + d8);
    #pragma unroll
    for (int j = 0; j < 8; ++j) qrow[d8 + j] = (float)qv[j];
  }
  float lg[20];
  float mx = -1e30f;
  for (int g = 0; g < 20; ++g) {
    float s = 0.f;
    const float* gp = gp_w + (size_t)h * 20 * HD + (size_t)g * HD;
    #pragma unroll
    for (int d4 = 0; d4 < HD; d4 += 4) {
      v4f gv = *(const v4f*)&gp[d4];
      s += qrow[d4] * gv[0] + qrow[d4 + 1] * gv[1] +
           qrow[d4 + 2] * gv[2] + qrow[d4 + 3] * gv[3];
    }
    lg[g] = s * invt;
    mx = fmaxf(mx, lg[g]);
  }
  float sum = 0.f;
  for (int g = 0; g < 20; ++g) { lg[g] = __expf(lg[g] - mx); sum += lg[g]; }
  const float inv = 1.0f / sum;
  for (int g = 0; g < 20; ++g) gw_pad[(size_t)idx * 32 + g] = (__bf16)(lg[g] * inv);
  for (int g = 20; g < 32; ++g) gw_pad[(size_t)idx * 32 + g] = (__bf16)0.0f;
}

__global__ void sga_zero(float* p, int n) {
  int i = blockIdx.x * 256 + threadIdx.x;
  if (i < n) p[i] = 0.f;
}

// ---------------------------------------------------------------------------
// Kernel 3: per (b,h, 16-query tile), single wave.
// ---------------------------------------------------------------------------
__global__ __launch_bounds__(32)
void sga_colsum(const __bf16* __restrict__ q, const __bf16* __restrict__ kmat,
                const __bf16* __restrict__ gw, float* __restrict__ rowstats,
                float* __restrict__ colsum) {
  const int blk = blockIdx.x;
  const int n0 = (blk & 63) * 16;
  const int bh = blk >> 6;
  const int lane = threadIdx.x, lhalf = lane >> 4, lmod = lane & 15;
  const size_t base = (size_t)bh * NN * HD;
  const float scale = 0.17677669529663687f;   // 32^-0.5

  const __bf16* qp = q  + base + (size_t)(n0 + lmod) * HD + lhalf * 16;
  const __bf16* gp = gw + base + (size_t)(n0 + lmod) * HD + lhalf * 16;
  v16bf aq = cat8(ldg8(qp), ldg8(qp + 8));
  v16bf ag = cat8(ldg8(gp), ldg8(gp + 8));

  float rmax[8], rsum[8];
  #pragma unroll
  for (int r = 0; r < 8; ++r) { rmax[r] = -1e30f; rsum[r] = 0.f; }

  for (int m0 = 0; m0 < NN; m0 += 16) {
    const __bf16* kp = kmat + base + (size_t)(m0 + lmod) * HD + lhalf * 16;
    v16bf bk = cat8(ldg8(kp), ldg8(kp + 8));
    v8f z = {};
    v8f s = wmma_bf16(aq, bk, z);
    #pragma unroll
    for (int r = 0; r < 8; ++r) {
      float v = s[r] * scale;
      float cm = v;
      cm = fmaxf(cm, __shfl_xor(cm, 1));
      cm = fmaxf(cm, __shfl_xor(cm, 2));
      cm = fmaxf(cm, __shfl_xor(cm, 4));
      cm = fmaxf(cm, __shfl_xor(cm, 8));
      const float nm = fmaxf(rmax[r], cm);
      float ev = __expf(v - nm);
      ev += __shfl_xor(ev, 1);
      ev += __shfl_xor(ev, 2);
      ev += __shfl_xor(ev, 4);
      ev += __shfl_xor(ev, 8);
      rsum[r] = rsum[r] * __expf(rmax[r] - nm) + ev;
      rmax[r] = nm;
    }
  }
  if (lmod == 0) {
    #pragma unroll
    for (int r = 0; r < 8; ++r) {
      const int n = n0 + r + lhalf * 8;
      rowstats[((size_t)bh * NN + n) * 2 + 0] = rmax[r];
      rowstats[((size_t)bh * NN + n) * 2 + 1] = rsum[r];
    }
  }
  for (int m0 = 0; m0 < NN; m0 += 16) {
    const __bf16* kp = kmat + base + (size_t)(m0 + lmod) * HD + lhalf * 16;
    const __bf16* gq = gw   + base + (size_t)(m0 + lmod) * HD + lhalf * 16;
    v16bf bk = cat8(ldg8(kp), ldg8(kp + 8));
    v16bf bg = cat8(ldg8(gq), ldg8(gq + 8));
    v8f z = {};
    v8f s = wmma_bf16(aq, bk, z);
    v8f g = wmma_bf16(ag, bg, z);
    float csum = 0.f;
    #pragma unroll
    for (int r = 0; r < 8; ++r)
      csum += __expf(s[r] * scale - rmax[r]) / rsum[r] * g[r];
    csum += __shfl_xor(csum, 16);   // rows 0-7 half + rows 8-15 half
    if (lhalf == 0)
      atomicAdd(&colsum[(size_t)bh * NN + m0 + lmod], csum);
  }
}

// ---------------------------------------------------------------------------
// Kernel 4: V' = V / (colsum + 1e-8), stored TRANSPOSED: vsT[bh][d][m]
// ---------------------------------------------------------------------------
__global__ __launch_bounds__(256)
void sga_vscale(const __bf16* __restrict__ v, const float* __restrict__ colsum,
                __bf16* __restrict__ vsT) {
  const int i = blockIdx.x * 256 + threadIdx.x;
  if (i >= BHT * NN * HD) return;
  const int bh = i >> 15;              // / (NN*HD)
  const int rest = i & (NN * HD - 1);
  const int m = rest >> 5, d = rest & 31;
  const float sc = 1.0f / (colsum[i >> 5] + 1e-8f);
  vsT[((size_t)bh * HD + d) * NN + m] = (__bf16)((float)v[i] * sc);
}

// ---------------------------------------------------------------------------
// Kernel 5: recompute P tiles, out_head = P @ V' via WMMA.
// ---------------------------------------------------------------------------
__global__ __launch_bounds__(32)
void sga_out(const __bf16* __restrict__ q, const __bf16* __restrict__ kmat,
             const __bf16* __restrict__ gw, const __bf16* __restrict__ vsT,
             const float* __restrict__ rowstats, __bf16* __restrict__ attn_out) {
  __shared__ __bf16 pbuf[16][LDP];
  const int blk = blockIdx.x;
  const int n0 = (blk & 63) * 16;
  const int bh = blk >> 6;
  const int lane = threadIdx.x, lhalf = lane >> 4, lmod = lane & 15;
  const size_t base  = (size_t)bh * NN * HD;
  const float scale = 0.17677669529663687f;

  const __bf16* qp = q  + base + (size_t)(n0 + lmod) * HD + lhalf * 16;
  const __bf16* gp = gw + base + (size_t)(n0 + lmod) * HD + lhalf * 16;
  v16bf aq = cat8(ldg8(qp), ldg8(qp + 8));
  v16bf ag = cat8(ldg8(gp), ldg8(gp + 8));

  float rmax[8], rsum[8];
  #pragma unroll
  for (int r = 0; r < 8; ++r) {
    const int n = n0 + r + lhalf * 8;
    rmax[r] = rowstats[((size_t)bh * NN + n) * 2 + 0];
    rsum[r] = rowstats[((size_t)bh * NN + n) * 2 + 1];
  }

  // V' row pointers for this lane's two output columns (d = lmod, 16+lmod)
  const __bf16* vrow0 = vsT + ((size_t)bh * HD + lmod) * NN;
  const __bf16* vrow1 = vsT + ((size_t)bh * HD + 16 + lmod) * NN;

  v8f acc0 = {}, acc1 = {};
  for (int mm = 0; mm < NN; mm += 32) {
    #pragma unroll
    for (int t = 0; t < 2; ++t) {
      const int m0 = mm + t * 16;
      const __bf16* kp = kmat + base + (size_t)(m0 + lmod) * HD + lhalf * 16;
      const __bf16* gq = gw   + base + (size_t)(m0 + lmod) * HD + lhalf * 16;
      v16bf bk = cat8(ldg8(kp), ldg8(kp + 8));
      v16bf bg = cat8(ldg8(gq), ldg8(gq + 8));
      v8f z = {};
      v8f s = wmma_bf16(aq, bk, z);
      v8f g = wmma_bf16(ag, bg, z);
      #pragma unroll
      for (int r = 0; r < 8; ++r)
        pbuf[r + lhalf * 8][t * 16 + lmod] =
            (__bf16)(__expf(s[r] * scale - rmax[r]) / rsum[r] * g[r]);
    }
    __syncthreads();
    v16bf pa = cat8(*(const v8bf*)&pbuf[lmod][lhalf * 16],
                    *(const v8bf*)&pbuf[lmod][lhalf * 16 + 8]);
    __syncthreads();
    // B = V' chunk (K = 32 keys contiguous per lane, N = output dim)
    const int koff = mm + lhalf * 16;
    v16bf bv0 = cat8(ldg8(vrow0 + koff), ldg8(vrow0 + koff + 8));
    v16bf bv1 = cat8(ldg8(vrow1 + koff), ldg8(vrow1 + koff + 8));
    acc0 = wmma_bf16(pa, bv0, acc0);
    acc1 = wmma_bf16(pa, bv1, acc1);
  }
  const int b = bh / NH, h = bh % NH;
  #pragma unroll
  for (int r = 0; r < 8; ++r) {
    const int n = n0 + r + lhalf * 8;
    const size_t o = ((size_t)b * NN + n) * DIMC + h * HD;
    attn_out[o + lmod]      = (__bf16)acc0[r];
    attn_out[o + 16 + lmod] = (__bf16)acc1[r];
  }
}

// ---------------------------------------------------------------------------
// Kernel 6: final projection attn_out(8192x384) @ W_proj(384x384) -> f32 out
// ---------------------------------------------------------------------------
__global__ __launch_bounds__(256)
void sga_proj_gemm(const __bf16* __restrict__ A, const float* __restrict__ W,
                   float* __restrict__ out) {
  __shared__ __bf16 xa[128][LDP];
  __shared__ __bf16 wbT[64][LDP];
  const int tid = threadIdx.x;
  const int wave = tid >> 5, lane = tid & 31;
  const int waveM = wave >> 1, waveN = wave & 1;
  const int bm = blockIdx.x * 128, bn = blockIdx.y * 64;
  const int lhalf = lane >> 4, lmod = lane & 15;
  v8f acc[2][2] = {};

  for (int k0 = 0; k0 < DIMC; k0 += 32) {
    // A tile: bf16 -> bf16, straight v8bf copies. 512 chunks of 8.
    #pragma unroll
    for (int it = 0; it < 2; ++it) {
      const int i = tid + it * 256;
      const int r = i >> 2, c8 = (i & 3) * 8;
      *(v8bf*)&xa[r][c8] = ldg8(&A[(size_t)(bm + r) * DIMC + k0 + c8]);
    }
    #pragma unroll
    for (int it = 0; it < 2; ++it) {
      const int i = tid + it * 256;
      const int r = i >> 4, c4 = (i & 15) * 4;
      v4f f = *(const v4f*)&W[(size_t)(k0 + r) * DIMC + bn + c4];
      wbT[c4 + 0][r] = (__bf16)f[0];
      wbT[c4 + 1][r] = (__bf16)f[1];
      wbT[c4 + 2][r] = (__bf16)f[2];
      wbT[c4 + 3][r] = (__bf16)f[3];
    }
    if (k0 + 32 < DIMC) {
      __builtin_prefetch(&A[(size_t)(bm + (tid >> 1)) * DIMC + k0 + 32], 0, 1);
      __builtin_prefetch(&W[(size_t)(k0 + 32 + (tid >> 3)) * DIMC + bn], 0, 1);
    }
    __syncthreads();
    #pragma unroll
    for (int tm = 0; tm < 2; ++tm) {
      const int row = waveM * 32 + tm * 16 + lmod;
      const int kb = lhalf * 16;
      v16bf a = cat8(*(const v8bf*)&xa[row][kb],
                     *(const v8bf*)&xa[row][kb + 8]);
      #pragma unroll
      for (int tn = 0; tn < 2; ++tn) {
        const int col = waveN * 32 + tn * 16 + lmod;
        v16bf b = cat8(*(const v8bf*)&wbT[col][lhalf * 16],
                       *(const v8bf*)&wbT[col][lhalf * 16 + 8]);
        acc[tm][tn] = wmma_bf16(a, b, acc[tm][tn]);
      }
    }
    __syncthreads();
  }
  #pragma unroll
  for (int tm = 0; tm < 2; ++tm)
    #pragma unroll
    for (int tn = 0; tn < 2; ++tn)
      #pragma unroll
      for (int r = 0; r < 8; ++r) {
        const int row = bm + waveM * 32 + tm * 16 + r + lhalf * 8;
        const int col = bn + waveN * 32 + tn * 16 + lmod;
        out[(size_t)row * DIMC + col] = acc[tm][tn][r];
      }
}

// ---------------------------------------------------------------------------
extern "C" void kernel_launch(void* const* d_in, const int* in_sizes, int n_in,
                              void* d_out, int out_size, void* d_ws, size_t ws_size,
                              hipStream_t stream) {
  const float* x      = (const float*)d_in[0];
  const float* W_qkv  = (const float*)d_in[1];
  const float* gp_w   = (const float*)d_in[2];
  const float* t_par  = (const float*)d_in[3];
  const float* W_proj = (const float*)d_in[4];
  float* out = (float*)d_out;

  // workspace carve-up (bytes)
  char* ws = (char*)d_ws;
  const size_t qkv_b = (size_t)BHT * NN * HD * sizeof(__bf16);   // 6 MiB each
  __bf16* q      = (__bf16*)ws;                 ws += qkv_b;
  __bf16* k      = (__bf16*)ws;                 ws += qkv_b;
  __bf16* v      = (__bf16*)ws;                 ws += qkv_b;
  __bf16* gwp    = (__bf16*)ws;                 ws += qkv_b;     // gw padded to 32
  float*  rowst  = (float*)ws;                  ws += (size_t)BHT * NN * 2 * sizeof(float);
  float*  colsum = (float*)ws;                  ws += (size_t)BHT * NN * sizeof(float);
  __bf16* vsT    = (__bf16*)ws;                 ws += qkv_b;     // transposed V'
  __bf16* aout   = (__bf16*)ws;                 ws += (size_t)ROWS * DIMC * sizeof(__bf16);
  (void)ws_size; (void)in_sizes; (void)n_in; (void)out_size;

  // 1) QKV projection
  dim3 g1(ROWS / 128, QKVC / 64);
  sga_qkv_gemm<<<g1, 256, 0, stream>>>(x, W_qkv, q, k, v);

  // 2) group weights
  sga_gw<<<(BHT * NN) / 256, 256, 0, stream>>>(q, gp_w, t_par, gwp);

  // 3) zero colsum, then colsum pass
  sga_zero<<<(BHT * NN + 255) / 256, 256, 0, stream>>>(colsum, BHT * NN);
  sga_colsum<<<BHT * (NN / 16), 32, 0, stream>>>(q, k, gwp, rowst, colsum);

  // 4) fold query-axis renorm into V (transposed output)
  sga_vscale<<<(BHT * NN * HD) / 256, 256, 0, stream>>>(v, colsum, vsT);

  // 5) P @ V'
  sga_out<<<BHT * (NN / 16), 32, 0, stream>>>(q, k, gwp, vsT, rowst, aout);

  // 6) output projection
  dim3 g6(ROWS / 128, DIMC / 64);
  sga_proj_gemm<<<g6, 256, 0, stream>>>(aout, W_proj, out);
}
